// TransformerModel_3453153706608
// MI455X (gfx1250) — compile-verified, matrix operand order checked by reading, not run on hardware
//
#include <hip/hip_runtime.h>
#include <hip/hip_bf16.h>

#define TSEQ   2048
#define NBATCH 4
#define DMODEL 512
#define NHEAD  8
#define DHEAD  64
#define DFFN   1024
#define NLAYER 6
#define NTOK   (NBATCH * TSEQ)   // 8192 rows

typedef __attribute__((ext_vector_type(16))) __bf16 v16bf;
typedef __attribute__((ext_vector_type(8)))  float  v8f;
typedef unsigned int v4u  __attribute__((ext_vector_type(4)));
typedef int          v8i_ __attribute__((ext_vector_type(8)));
typedef int          v4i_ __attribute__((ext_vector_type(4)));

union Frag16 { v16bf v; unsigned short u[16]; uint4 q[2]; };
union AccF   { v8f v; float f[8]; };

__device__ inline unsigned short f2bf(float f) {
  unsigned u = __float_as_uint(f);
  u += 0x7FFFu + ((u >> 16) & 1u);      // round-to-nearest-even
  return (unsigned short)(u >> 16);
}

// Load one WMMA A/B fragment (16 rows x 32 K of bf16) for this lane.
// rowptr points at element [row_of_lane][kbase], kbase = (lane<16 ? 0 : 8).
__device__ inline v16bf load_frag(const unsigned short* rowptr) {
  Frag16 f;
  f.q[0] = *(const uint4*)(rowptr);
  f.q[1] = *(const uint4*)(rowptr + 16);
  return f.v;
}

__device__ inline v8f wmma_bf16(v16bf a, v16bf b, v8f c) {
  return __builtin_amdgcn_wmma_f32_16x16x32_bf16(
      /*neg_a=*/false, a, /*neg_b=*/false, b,
      /*c_mod=*/(short)0, c, /*reuse_a=*/false, /*reuse_b=*/false);
}

__device__ inline v8f zero_acc() {
  AccF z;
  #pragma unroll
  for (int i = 0; i < 8; ++i) z.f[i] = 0.0f;
  return z.v;
}

// ---------------------------------------------------------------- TDM
// 2D tensor tile load Global -> LDS via Tensor Data Mover.
// D# packing per CDNA5 ISA 8.3/8.4: group0 = {count/type/lds/global addr},
// group1 = {data_size, tensor dims, tile dims, strides}. data_size=1 -> 2B.
// This toolchain exposes the 6-arg builtin: (g0, g1, g2, g3, g4, cpol).
__device__ inline void tdm_load_2d_bf16(unsigned lds_addr, const void* gptr,
                                        unsigned tile_d0, unsigned tile_d1,
                                        unsigned tensor_d0, unsigned tensor_d1,
                                        unsigned long long stride0_elems) {
  const unsigned long long ga = (unsigned long long)gptr;
  v4u g0;
  g0.x = 1u;                                        // count=1, user descriptor
  g0.y = lds_addr;                                  // LDS byte address
  g0.z = (unsigned)(ga & 0xFFFFFFFFu);              // global_addr[31:0]
  g0.w = (unsigned)((ga >> 32) & 0x1FFFFFFu)        // global_addr[56:32]
         | (2u << 30);                              // type = 2 ("image")
  v8i_ g1;
  g1[0] = (int)(1u << 16);                          // data_size=1 (2 bytes)
  g1[1] = (int)((tensor_d0 & 0xFFFFu) << 16);       // tensor_dim0[15:0]
  g1[2] = (int)((tensor_d0 >> 16) | ((tensor_d1 & 0xFFFFu) << 16));
  g1[3] = (int)((tensor_d1 >> 16) | (tile_d0 << 16));   // tile_dim0
  g1[4] = (int)(tile_d1 & 0xFFFFu);                 // tile_dim1 (tile_dim2=0)
  g1[5] = (int)(stride0_elems & 0xFFFFFFFFull);     // tensor_dim0_stride lo
  g1[6] = (int)((stride0_elems >> 32) & 0xFFFFull); // stride hi, dim1_stride=0
  g1[7] = 0;
  v4i_ g2 = {0, 0, 0, 0};
  v4i_ g3 = {0, 0, 0, 0};
  v8i_ g4 = {0, 0, 0, 0, 0, 0, 0, 0};
  __builtin_amdgcn_tensor_load_to_lds(g0, g1, g2, g3, g4, 0);
}

// ---------------------------------------------------------------- convert
__global__ __launch_bounds__(256) void convert_f32_bf16(
    const float* __restrict__ src, unsigned short* __restrict__ dst,
    long long count) {
  long long i = (long long)blockIdx.x * blockDim.x + threadIdx.x;
  long long stride = (long long)gridDim.x * blockDim.x;
  for (; i < count; i += stride) dst[i] = f2bf(src[i]);
}

// ---------------------------------------------------------------- embed+pos
__global__ __launch_bounds__(256) void embed_kernel(
    const int* __restrict__ x, const float* __restrict__ emb,
    float* __restrict__ h, unsigned short* __restrict__ hb) {
  const int row = blockIdx.x;           // 0..NTOK-1
  const int t   = row & (TSEQ - 1);
  const int tok = x[row];
  const int d0  = threadIdx.x * 2;
  #pragma unroll
  for (int j = 0; j < 2; ++j) {
    const int d = d0 + j;
    const float ex   = (float)(2 * (d >> 1)) / (float)DMODEL;
    const float rate = powf(10000.0f, -ex);
    const float ang  = (float)t * rate;
    const float pe   = (d & 1) ? cosf(ang) : sinf(ang);
    const float val  = emb[(size_t)tok * DMODEL + d] * 22.62741699796952f + pe;
    h[(size_t)row * DMODEL + d]  = val;
    hb[(size_t)row * DMODEL + d] = f2bf(val);
  }
}

// ---------------------------------------------------------------- GEMM
// C[M,N] = A[M,K](bf16) * B[K,N](bf16) + bias, fused epilogue.
// A tile staged by the Tensor Data Mover (TENSORcnt); B tile via register
// transpose so WMMA B-fragments are contiguous 16B LDS reads.
// mode 0: QKV split -> bf16 [N,H,T,DH]; mode 1: ReLU -> bf16; mode 2: f32.
__global__ __launch_bounds__(256) void gemm_bf16_kernel(
    const unsigned short* __restrict__ A,
    const unsigned short* __restrict__ B,
    const float* __restrict__ bias,
    int M, int N, int K, int mode,
    unsigned short* __restrict__ outB,
    float* __restrict__ outF,
    unsigned short* __restrict__ Qd,
    unsigned short* __restrict__ Kd,
    unsigned short* __restrict__ Vd) {
  __shared__ unsigned short lA[128 * 32];   // A tile, row-major [128][32]
  __shared__ unsigned short lBt[64 * 32];   // B tile transposed [64(N)][32(K)]

  const int tid     = threadIdx.x;
  const int lane    = tid & 31;
  const int wave    = tid >> 5;          // 8 waves
  const int m0      = (wave >> 1) * 32;  // wave tile origin in block
  const int n0      = (wave & 1) * 32;
  const int bM      = blockIdx.x * 128;
  const int bN      = blockIdx.y * 64;
  const int laneRow = lane & 15;
  const int kb      = (lane < 16) ? 0 : 8;
  const int halfOff = (lane < 16) ? 0 : 8;

  AccF acc[2][2];
  #pragma unroll
  for (int i = 0; i < 2; ++i)
    #pragma unroll
    for (int j = 0; j < 2; ++j) acc[i][j].v = zero_acc();

  const int br = tid >> 3;            // 0..31  (K row)
  const int bc = (tid & 7) * 8;       // 0..56  (N col)
  const unsigned ldsA = (unsigned)(size_t)&lA[0];

  for (int k0 = 0; k0 < K; k0 += 32) {
    const uint4 bv = *(const uint4*)(B + (size_t)(k0 + br) * N + bN + bc);
    if (k0 + 32 < K)
      __builtin_prefetch(B + (size_t)(k0 + 32 + br) * N + bN + bc, 0, 0);
    __syncthreads();                   // previous tile fully consumed
    if (tid < 32) {
      // DMA the 128x32 A tile straight into LDS (row-major, stride K).
      tdm_load_2d_bf16(ldsA, A + (size_t)bM * K + k0,
                       /*tile_d0=*/32, /*tile_d1=*/128,
                       /*tensor_d0=*/(unsigned)K, /*tensor_d1=*/(unsigned)M,
                       /*stride0=*/(unsigned long long)K);
      __builtin_amdgcn_s_wait_tensorcnt(0);
    }
    {
      const unsigned short* bs = (const unsigned short*)&bv;
      #pragma unroll
      for (int i = 0; i < 8; ++i) lBt[(bc + i) * 32 + br] = bs[i];
    }
    __syncthreads();

    const v16bf af0 = load_frag(lA + (m0 + laneRow) * 32 + kb);
    const v16bf af1 = load_frag(lA + (m0 + 16 + laneRow) * 32 + kb);
    const v16bf bf0 = load_frag(lBt + (n0 + laneRow) * 32 + kb);
    const v16bf bf1 = load_frag(lBt + (n0 + 16 + laneRow) * 32 + kb);
    acc[0][0].v = wmma_bf16(af0, bf0, acc[0][0].v);
    acc[0][1].v = wmma_bf16(af0, bf1, acc[0][1].v);
    acc[1][0].v = wmma_bf16(af1, bf0, acc[1][0].v);
    acc[1][1].v = wmma_bf16(af1, bf1, acc[1][1].v);
  }

  #pragma unroll
  for (int i = 0; i < 2; ++i) {
    #pragma unroll
    for (int j = 0; j < 2; ++j) {
      #pragma unroll
      for (int v = 0; v < 8; ++v) {
        const int row = bM + m0 + i * 16 + v + halfOff;
        const int col = bN + n0 + j * 16 + laneRow;
        const float val = acc[i][j].f[v] + bias[col];
        if (mode == 0) {
          // col = head*192 + dh*3 + c  (reshape (H, DH, 3))
          const int head = col / 192;
          const int rem  = col % 192;
          const int dh   = rem / 3;
          const int c    = rem % 3;
          const int t    = row & (TSEQ - 1);
          const int nb   = row >> 11;   // row / TSEQ
          const size_t idx =
              (((size_t)nb * NHEAD + head) * TSEQ + t) * DHEAD + dh;
          const unsigned short bvv = f2bf(val);
          if (c == 0)      Qd[idx] = bvv;
          else if (c == 1) Kd[idx] = bvv;
          else             Vd[idx] = bvv;
        } else if (mode == 1) {
          outB[(size_t)row * N + col] = f2bf(fmaxf(val, 0.0f));
        } else {
          outF[(size_t)row * N + col] = val;
        }
      }
    }
  }
}

// ---------------------------------------------------------------- attention
// Flash attention, causal + pad mask. Block: 8 waves, one (n,h) pair,
// 128 queries (16 per wave). Keys processed 32 at a time. V tile staged via
// async global->LDS copy (ASYNCcnt).
__global__ __launch_bounds__(256) void attn_kernel(
    const unsigned short* __restrict__ Qm,
    const unsigned short* __restrict__ Km,
    const unsigned short* __restrict__ Vm,
    const int* __restrict__ x,
    float* __restrict__ out) {
  __shared__ unsigned short sV[32 * DHEAD];    // raw V tile [key][dh]
  __shared__ unsigned short sP[8][16 * 32];    // per-wave P tile [m][key]

  const int tid     = threadIdx.x;
  const int lane    = tid & 31;
  const int wave    = tid >> 5;
  const int qb      = blockIdx.x;
  const int h       = blockIdx.y;
  const int n       = blockIdx.z;
  const int t0      = qb * 128 + wave * 16;
  const int laneRow = lane & 15;
  const int kb      = (lane < 16) ? 0 : 8;
  const int halfOff = (lane < 16) ? 0 : 8;

  const size_t headBase = ((size_t)n * NHEAD + h) * TSEQ * DHEAD;
  const unsigned short* Qh = Qm + headBase;
  const unsigned short* Kh = Km + headBase;
  const unsigned short* Vh = Vm + headBase;
  const int* xr = x + n * TSEQ;

  // Q fragments: 16 queries x 64 dh = two K-chunks
  const v16bf aQ0 = load_frag(Qh + (size_t)(t0 + laneRow) * DHEAD + kb);
  const v16bf aQ1 = load_frag(Qh + (size_t)(t0 + laneRow) * DHEAD + 32 + kb);

  float mrow[8], lrow[8];
  AccF O[4];
  #pragma unroll
  for (int v = 0; v < 8; ++v) { mrow[v] = -3.0e38f; lrow[v] = 0.0f; }
  #pragma unroll
  for (int c = 0; c < 4; ++c) O[c].v = zero_acc();

  const int vr = tid >> 3;        // key within tile 0..31
  const int vc = (tid & 7) * 8;   // dh base
  const unsigned ldsV =
      (unsigned)(size_t)&sV[0] + (unsigned)((vr * DHEAD + vc) * 2);

  const int sEnd = (qb + 1) * 128;   // causal upper bound for this block
  for (int s0 = 0; s0 < sEnd; s0 += 32) {
    __syncthreads();                 // previous V tile fully consumed
    // async copy: 16B per lane, global -> LDS, no VGPR round trip
    {
      const unsigned long long ga =
          (unsigned long long)(Vh + (size_t)(s0 + vr) * DHEAD + vc);
      asm volatile("global_load_async_to_lds_b128 %0, %1, off"
                   :: "v"(ldsV), "v"(ga)
                   : "memory");
      asm volatile("s_wait_asynccnt 0" ::: "memory");
    }
    __syncthreads();

    // S = Q K^T  (16 queries x 32 keys), dh reduced over two WMMA chunks
    AccF S[2];
    #pragma unroll
    for (int sub = 0; sub < 2; ++sub) {
      const unsigned short* kp =
          Kh + (size_t)(s0 + sub * 16 + laneRow) * DHEAD;
      const v16bf bk0 = load_frag(kp + kb);
      const v16bf bk1 = load_frag(kp + 32 + kb);
      const v8f tacc = wmma_bf16(aQ0, bk0, zero_acc());
      S[sub].v = wmma_bf16(aQ1, bk1, tacc);
    }

    // online softmax over the 32 keys
    unsigned short* Pw = sP[wave];
    #pragma unroll
    for (int v = 0; v < 8; ++v) {
      const int trow = t0 + v + halfOff;
      const int key0 = s0 + laneRow;
      const int key1 = key0 + 16;
      const float mk0 = ((xr[key0] == 0) ? 1.0f : 0.0f) +
                        ((trow < key0) ? 1.0f : 0.0f);
      const float mk1 = ((xr[key1] == 0) ? 1.0f : 0.0f) +
                        ((trow < key1) ? 1.0f : 0.0f);
      float s0v = S[0].f[v] * 0.125f - 1.0e9f * mk0;
      float s1v = S[1].f[v] * 0.125f - 1.0e9f * mk1;
      float rm = fmaxf(s0v, s1v);
      #pragma unroll
      for (int off = 1; off < 16; off <<= 1)
        rm = fmaxf(rm, __shfl_xor(rm, off, 16));
      const float mn = fmaxf(mrow[v], rm);
      const float p0 = __expf(s0v - mn);
      const float p1 = __expf(s1v - mn);
      float rs = p0 + p1;
      #pragma unroll
      for (int off = 1; off < 16; off <<= 1) rs += __shfl_xor(rs, off, 16);
      const float corr = __expf(mrow[v] - mn);
      lrow[v] = lrow[v] * corr + rs;
      mrow[v] = mn;
      #pragma unroll
      for (int c = 0; c < 4; ++c) O[c].f[v] *= corr;
      const int prow = v + halfOff;
      Pw[prow * 32 + laneRow]      = f2bf(p0);
      Pw[prow * 32 + laneRow + 16] = f2bf(p1);
    }

    // O += P V  (K=32 keys, 4 dh-chunks of 16); V read strided from raw tile
    const v16bf aP = load_frag(Pw + laneRow * 32 + kb);
    #pragma unroll
    for (int c = 0; c < 4; ++c) {
      const int dh = c * 16 + laneRow;
      Frag16 fv;
      #pragma unroll
      for (int i = 0; i < 8; ++i) {
        fv.u[i]     = sV[(kb + i) * DHEAD + dh];
        fv.u[i + 8] = sV[(kb + 16 + i) * DHEAD + dh];
      }
      O[c].v = wmma_bf16(aP, fv.v, O[c].v);
    }
  }

  // finalize + write [N,T,D]
  #pragma unroll
  for (int c = 0; c < 4; ++c) {
    #pragma unroll
    for (int v = 0; v < 8; ++v) {
      const int trow = t0 + v + halfOff;
      const int dh   = c * 16 + laneRow;
      out[((size_t)n * TSEQ + trow) * DMODEL + h * DHEAD + dh] =
          O[c].f[v] / fmaxf(lrow[v], 1e-20f);
    }
  }
}

// ---------------------------------------------------------------- add + LN
__global__ __launch_bounds__(256) void add_ln_kernel(
    const float* __restrict__ hin, const float* __restrict__ res,
    const float* __restrict__ g, const float* __restrict__ b,
    float* __restrict__ hout, unsigned short* __restrict__ hbout) {
  __shared__ float red[16];
  const int row  = blockIdx.x;
  const int tid  = threadIdx.x;
  const int lane = tid & 31;
  const int wave = tid >> 5;

  const float2 hv = ((const float2*)(hin + (size_t)row * DMODEL))[tid];
  const float2 rv = ((const float2*)(res + (size_t)row * DMODEL))[tid];
  const float a0 = hv.x + rv.x;
  const float a1 = hv.y + rv.y;
  float s = a0 + a1;
  float q = a0 * a0 + a1 * a1;
  #pragma unroll
  for (int off = 16; off; off >>= 1) {
    s += __shfl_xor(s, off, 32);
    q += __shfl_xor(q, off, 32);
  }
  if (lane == 0) { red[wave] = s; red[8 + wave] = q; }
  __syncthreads();
  float S = 0.0f, Q = 0.0f;
  #pragma unroll
  for (int wv = 0; wv < 8; ++wv) { S += red[wv]; Q += red[8 + wv]; }
  const float mu  = S * (1.0f / DMODEL);
  const float var = Q * (1.0f / DMODEL) - mu * mu;
  const float inv = rsqrtf(var + 1e-3f);

  const int d0 = tid * 2;
  const float y0 = (a0 - mu) * inv * g[d0] + b[d0];
  const float y1 = (a1 - mu) * inv * g[d0 + 1] + b[d0 + 1];
  float* op = hout + (size_t)row * DMODEL;
  op[d0] = y0;
  op[d0 + 1] = y1;
  unsigned short* ob = hbout + (size_t)row * DMODEL;
  ob[d0] = f2bf(y0);
  ob[d0 + 1] = f2bf(y1);
}

// ---------------------------------------------------------------- host
extern "C" void kernel_launch(void* const* d_in, const int* in_sizes, int n_in,
                              void* d_out, int out_size, void* d_ws,
                              size_t ws_size, hipStream_t stream) {
  const int*   x    = (const int*)d_in[0];
  const float* emb  = (const float*)d_in[1];
  const float* Wqkv = (const float*)d_in[2];
  const float* bqkv = (const float*)d_in[3];
  const float* Wff  = (const float*)d_in[4];
  const float* bff  = (const float*)d_in[5];
  const float* Wo   = (const float*)d_in[6];
  const float* bo   = (const float*)d_in[7];
  const float* ln1g = (const float*)d_in[8];
  const float* ln1b = (const float*)d_in[9];
  const float* ln2g = (const float*)d_in[10];
  const float* ln2b = (const float*)d_in[11];

  char* w = (char*)d_ws;
  size_t off = 0;
  auto take = [&](size_t bytes) -> void* {
    void* p = w + off;
    off += (bytes + 255) & ~(size_t)255;
    return p;
  };

  float*          hbuf  = (float*)take((size_t)NTOK * DMODEL * 4);
  unsigned short* hb16  = (unsigned short*)take((size_t)NTOK * DMODEL * 2);
  unsigned short* Qb    = (unsigned short*)take((size_t)NTOK * DMODEL * 2);
  unsigned short* Kb    = (unsigned short*)take((size_t)NTOK * DMODEL * 2);
  unsigned short* Vb    = (unsigned short*)take((size_t)NTOK * DMODEL * 2);
  float*          attnb = (float*)take((size_t)NTOK * DMODEL * 4);
  unsigned short* ff1b  = (unsigned short*)take((size_t)NTOK * DFFN * 2);
  float*          ffout = (float*)take((size_t)NTOK * DMODEL * 4);

  const long long nWqkv = (long long)NLAYER * DMODEL * (3 * DMODEL);
  const long long nWff  = (long long)NLAYER * DMODEL * DFFN;
  const long long nWo   = (long long)NLAYER * DFFN * DMODEL;
  unsigned short* WqkvB = (unsigned short*)take((size_t)nWqkv * 2);
  unsigned short* WffB  = (unsigned short*)take((size_t)nWff * 2);
  unsigned short* WoB   = (unsigned short*)take((size_t)nWo * 2);

  convert_f32_bf16<<<2048, 256, 0, stream>>>(Wqkv, WqkvB, nWqkv);
  convert_f32_bf16<<<2048, 256, 0, stream>>>(Wff, WffB, nWff);
  convert_f32_bf16<<<2048, 256, 0, stream>>>(Wo, WoB, nWo);

  embed_kernel<<<NTOK, 256, 0, stream>>>(x, emb, hbuf, hb16);

  for (int l = 0; l < NLAYER; ++l) {
    // QKV projection (fused split into Q/K/V bf16 [N,H,T,DH])
    gemm_bf16_kernel<<<dim3(NTOK / 128, (3 * DMODEL) / 64), 256, 0, stream>>>(
        hb16, WqkvB + (size_t)l * DMODEL * 3 * DMODEL,
        bqkv + (size_t)l * 3 * DMODEL, NTOK, 3 * DMODEL, DMODEL, 0,
        nullptr, nullptr, Qb, Kb, Vb);
    // Flash attention
    attn_kernel<<<dim3(TSEQ / 128, NHEAD, NBATCH), 256, 0, stream>>>(
        Qb, Kb, Vb, x, attnb);
    // residual + LN1
    add_ln_kernel<<<NTOK, 256, 0, stream>>>(
        hbuf, attnb, ln1g + (size_t)l * DMODEL, ln1b + (size_t)l * DMODEL,
        hbuf, hb16);
    // FFN up (ReLU fused)
    gemm_bf16_kernel<<<dim3(NTOK / 128, DFFN / 64), 256, 0, stream>>>(
        hb16, WffB + (size_t)l * DMODEL * DFFN, bff + (size_t)l * DFFN,
        NTOK, DFFN, DMODEL, 1, ff1b, nullptr, nullptr, nullptr, nullptr);
    // FFN down
    gemm_bf16_kernel<<<dim3(NTOK / 128, DMODEL / 64), 256, 0, stream>>>(
        ff1b, WoB + (size_t)l * DFFN * DMODEL, bo + (size_t)l * DMODEL,
        NTOK, DMODEL, DFFN, 2, nullptr, ffout, nullptr, nullptr, nullptr);
    // residual + LN2
    add_ln_kernel<<<NTOK, 256, 0, stream>>>(
        hbuf, ffout, ln2g + (size_t)l * DMODEL, ln2b + (size_t)l * DMODEL,
        hbuf, hb16);
  }

  (void)hipMemcpyAsync(d_out, hbuf, (size_t)NTOK * DMODEL * 4,
                       hipMemcpyDeviceToDevice, stream);
}
